// WindowShiftAttention2_38491496907143
// MI455X (gfx1250) — matrix operand order repaired and checked by reading, not compiled, not run on hardware
//
#include <hip/hip_runtime.h>
#include <hip/hip_bf16.h>

typedef __attribute__((ext_vector_type(16))) __bf16 v16bf;
typedef __attribute__((ext_vector_type(8)))  __bf16 v8bf;
typedef __attribute__((ext_vector_type(4)))  __bf16 v4bf;
typedef __attribute__((ext_vector_type(8)))  float  v8f;
typedef __attribute__((ext_vector_type(4)))  float  v4f;

#define BB    8
#define LL    6000
#define CC    256
#define HH    8
#define WW    300
#define DD    32
#define NW    20          // L / W
#define BW    160         // B * NW
#define ROWS  48000       // BW * W
#define WP    304         // q/k rows padded to 19*16
#define VP    320         // v rows padded to 10*32
#define SCALE 0.17677669529663687f  // 1/sqrt(32)

// ---------------- helpers ----------------

__device__ inline v16bf pack16(v8bf lo, v8bf hi) {
    v16bf a;
#pragma unroll
    for (int i = 0; i < 8; ++i) { a[i] = lo[i]; a[i + 8] = hi[i]; }
    return a;
}

// A-fragment (16x32 bf16, M x K): lane l holds row M=l&15;
// elements e=0..15 -> K = 16*g + e, g = l>>4.
__device__ inline v16bf load_afrag(const __bf16* rowptr, int g) {
    v8bf lo = *(const v8bf*)(rowptr + 16 * g);
    v8bf hi = *(const v8bf*)(rowptr + 16 * g + 8);
    return pack16(lo, hi);
}

__device__ inline v8f wmma_bf16(v16bf a, v16bf b, v8f c) {
    return __builtin_amdgcn_wmma_f32_16x16x32_bf16(false, a, false, b, (short)0, c,
                                                   false, false);
}

// shifted source position: t<W -> t ; else W + ((t-W+W/2) mod (L-W))
__device__ inline int shift_map(int ts, int shift) {
    return (!shift || ts < WW) ? ts : WW + (ts - WW + WW / 2) % (LL - WW);
}

// ---------------- kernel 1: shift + x+pos, fp32 -> bf16 ----------------

__global__ __launch_bounds__(256) void prep(const float* __restrict__ q,
                                            const float* __restrict__ qp,
                                            const int* __restrict__ shiftp,
                                            __bf16* __restrict__ xpos,
                                            __bf16* __restrict__ xbf) {
    int tid = blockIdx.x * 256 + threadIdx.x;        // ROWS*64 threads, 4 floats each
    int r  = tid >> 6;
    int c4 = (tid & 63) * 4;
    int b  = r / LL, ts = r % LL;
    int st = shift_map(ts, shiftp[0]);
    size_t src = ((size_t)(b * LL + st)) * CC + c4;
    v4f qv = *(const v4f*)(q + src);
    v4f pv = *(const v4f*)(qp + src);
    v4bf xb, xp;
#pragma unroll
    for (int i = 0; i < 4; ++i) {
        xb[i] = (__bf16)qv[i];
        xp[i] = (__bf16)(qv[i] + pv[i]);
    }
    size_t dst = (size_t)r * CC + c4;
    *(v4bf*)(xbf + dst)  = xb;
    *(v4bf*)(xpos + dst) = xp;
}

// ---------------- kernel 2: weights -> bf16, transposed (col-major) ----------------
// wT[c][k]: c in [0,512) from qk_w, [512,768) from v_w ; pT[c][k] from proj_w.

__global__ __launch_bounds__(128) void wprep(const float* __restrict__ qk_w,
                                             const float* __restrict__ v_w,
                                             const float* __restrict__ proj_w,
                                             __bf16* __restrict__ wT,
                                             __bf16* __restrict__ pT) {
    int c = blockIdx.x;                               // 0..1023
    for (int k = threadIdx.x; k < CC; k += 128) {
        float v;
        __bf16* dst;
        if (c < 512)      { v = qk_w[k * 512 + c];          dst = wT + (size_t)c * CC; }
        else if (c < 768) { v = v_w[k * CC + (c - 512)];    dst = wT + (size_t)c * CC; }
        else              { v = proj_w[k * CC + (c - 768)]; dst = pT + (size_t)(c - 768) * CC; }
        dst[k] = (__bf16)v;
    }
}

// ---------------- kernel 3: zero q/k/v padding rows ----------------

__global__ __launch_bounds__(128) void padzero(__bf16* __restrict__ qw,
                                               __bf16* __restrict__ kw,
                                               __bf16* __restrict__ vw) {
    int bwh = blockIdx.x;                             // 0..1279
    __bf16 z = (__bf16)0.0f;
    __bf16* qp = qw + (size_t)bwh * WP * DD + WW * DD;
    __bf16* kp = kw + (size_t)bwh * WP * DD + WW * DD;
    __bf16* vp = vw + (size_t)bwh * VP * DD + WW * DD;
    int t = threadIdx.x;
    if (t < (WP - WW) * DD) { qp[t] = z; kp[t] = z; }
    for (int i = t; i < (VP - WW) * DD; i += 128) vp[i] = z;
}

// ---------------- kernel 4: fused QKV GEMM [48000 x 256] @ [256 x 768] ----------------
// one wave computes a 32x64 output tile: 2 A-frags x 4 B-frags = 8 WMMAs / K-chunk.
// 64-col blocks: ct4 0..3 -> q (scaled), 4..7 -> k, 8..11 -> v (A switches to xbf).

__global__ __launch_bounds__(32) void gemm_qkv(const __bf16* __restrict__ xpos,
                                               const __bf16* __restrict__ xbf,
                                               const __bf16* __restrict__ wT,
                                               const float* __restrict__ qk_b,
                                               const float* __restrict__ v_b,
                                               __bf16* __restrict__ qw,
                                               __bf16* __restrict__ kw,
                                               __bf16* __restrict__ vw) {
    int ct4 = blockIdx.x;                 // 0..11  (64 cols each)
    int rt2 = blockIdx.y;                 // 0..1499 (32 rows each)
    int l = threadIdx.x, g = l >> 4, n = l & 15;
    const __bf16* A   = (ct4 < 8) ? xpos : xbf;
    const __bf16* ap0 = A + (size_t)(rt2 * 32 + n) * CC;
    const __bf16* ap1 = ap0 + 16 * CC;
    const __bf16* bp  = wT + (size_t)(ct4 * 64 + n) * CC;

    v8f acc[2][4] = {};
#pragma unroll
    for (int kc = 0; kc < 8; ++kc) {
        v16bf a0 = load_afrag(ap0 + kc * 32, g);
        v16bf a1 = load_afrag(ap1 + kc * 32, g);
#pragma unroll
        for (int i = 0; i < 4; ++i) {
            v16bf b = *(const v16bf*)(bp + (size_t)i * 16 * CC + kc * 32 + 16 * g);
            acc[0][i] = wmma_bf16(a0, b, acc[0][i]);
            acc[1][i] = wmma_bf16(a1, b, acc[1][i]);
        }
    }

#pragma unroll
    for (int i = 0; i < 4; ++i) {
        int c = ct4 * 64 + i * 16 + n;
        float bias, scl = 1.0f;
        __bf16* dst;
        int hd, rpad;
        if (c < 256)      { bias = qk_b[c];      scl = SCALE; dst = qw; hd = c;       rpad = WP; }
        else if (c < 512) { bias = qk_b[c];                   dst = kw; hd = c - 256; rpad = WP; }
        else              { bias = v_b[c - 512];              dst = vw; hd = c - 512; rpad = VP; }
        int h = hd >> 5, dd = hd & 31;
#pragma unroll
        for (int s = 0; s < 2; ++s)
#pragma unroll
            for (int r = 0; r < 8; ++r) {
                int row = rt2 * 32 + s * 16 + r + 8 * g;     // 0..47999
                int bw = row / WW, wr = row % WW;
                float val = (acc[s][i][r] + bias) * scl;
                dst[(((size_t)(bw * HH + h)) * rpad + wr) * DD + dd] = (__bf16)val;
            }
    }
}

// ---------------- kernel 5: windowed attention, one block per (window, head) ----------------

#define WAVES_C 4

__global__ __launch_bounds__(128) void attn_win(const __bf16* __restrict__ qw,
                                                const __bf16* __restrict__ kw,
                                                const __bf16* __restrict__ vw,
                                                __bf16* __restrict__ ow) {
    __shared__ __bf16 kk[WP * DD];          // k rows       (19456 B)
    __shared__ __bf16 vT[DD * VP];          // v transposed (20480 B)
    __shared__ __bf16 pl[WAVES_C * 16 * VP];// per-wave P   (40960 B)

    int bwh = blockIdx.x;                   // 0..1279
    int bw = bwh >> 3, h = bwh & 7;
    int tid = threadIdx.x, wv = tid >> 5, l = tid & 31, g = l >> 4, n = l & 15;

    const __bf16* qs = qw + (size_t)bwh * WP * DD;
    const __bf16* ks = kw + (size_t)bwh * WP * DD;
    const __bf16* vs = vw + (size_t)bwh * VP * DD;

    // stage k (plain copy, dword granularity)
    {
        const unsigned* s = (const unsigned*)ks;
        unsigned* d = (unsigned*)kk;
        for (int i = tid; i < WP * DD / 2; i += 128) d[i] = s[i];
    }
    // stage v transposed: vT[dd][row] = v[row][dd]
    for (int i = tid; i < VP * DD; i += 128) {
        int row = i >> 5, dd = i & 31;
        vT[dd * VP + row] = vs[i];
    }
    // zero P panes once (cols >= 304 stay exactly zero for the padded K chunk)
    {
        unsigned* d = (unsigned*)pl;
        for (int i = tid; i < WAVES_C * 16 * VP / 2; i += 128) d[i] = 0u;
    }
    __syncthreads();

    __bf16* pw = pl + wv * 16 * VP;

    for (int t = wv; t < 19; t += WAVES_C) {          // q-row tiles (uniform per wave)
        // ---- scores: S = q_tile(16x32) @ k^T(32x16) per k tile ----
        v16bf aq = load_afrag(qs + (size_t)(t * 16 + n) * DD, g);
        float acc[19][8];
#pragma unroll
        for (int j = 0; j < 19; ++j) {
            v16bf b = *(const v16bf*)(kk + (j * 16 + n) * DD + 16 * g);
            v8f c = {};
            c = wmma_bf16(aq, b, c);
#pragma unroll
            for (int r = 0; r < 8; ++r) acc[j][r] = c[r];
        }
        // mask invalid k columns 300..303 (tile 18, N >= 12)
        if (n >= 12) {
#pragma unroll
            for (int r = 0; r < 8; ++r) acc[18][r] = -__builtin_inff();
        }
        // ---- softmax over 304 cols: C layout M=r+8g, N=(l&15)+16j ----
        float mrow[8], srow[8];
#pragma unroll
        for (int r = 0; r < 8; ++r) mrow[r] = -__builtin_inff();
#pragma unroll
        for (int j = 0; j < 19; ++j)
#pragma unroll
            for (int r = 0; r < 8; ++r) mrow[r] = fmaxf(mrow[r], acc[j][r]);
#pragma unroll
        for (int m = 1; m <= 8; m <<= 1)
#pragma unroll
            for (int r = 0; r < 8; ++r) mrow[r] = fmaxf(mrow[r], __shfl_xor(mrow[r], m, 32));
#pragma unroll
        for (int r = 0; r < 8; ++r) srow[r] = 0.0f;
#pragma unroll
        for (int j = 0; j < 19; ++j)
#pragma unroll
            for (int r = 0; r < 8; ++r) {
                acc[j][r] = __expf(acc[j][r] - mrow[r]);
                srow[r] += acc[j][r];
            }
#pragma unroll
        for (int m = 1; m <= 8; m <<= 1)
#pragma unroll
            for (int r = 0; r < 8; ++r) srow[r] += __shfl_xor(srow[r], m, 32);
#pragma unroll
        for (int r = 0; r < 8; ++r) srow[r] = 1.0f / srow[r];
        // write normalized P (bf16) into this wave's pane: P[M][col]
#pragma unroll
        for (int j = 0; j < 19; ++j)
#pragma unroll
            for (int r = 0; r < 8; ++r)
                pw[(r + 8 * g) * VP + n + 16 * j] = (__bf16)(acc[j][r] * srow[r]);

        // ---- out = P(16x320) @ v(320x32) : 2 N-halves x 10 K-chunks ----
#pragma unroll
        for (int nh = 0; nh < 2; ++nh) {
            v8f o = {};
#pragma unroll
            for (int kc = 0; kc < 10; ++kc) {
                v16bf pa = load_afrag(pw + n * VP + kc * 32, g);  // row M = l&15
                v16bf b  = *(const v16bf*)(vT + (nh * 16 + n) * VP + kc * 32 + 16 * g);
                o = wmma_bf16(pa, b, o);
            }
#pragma unroll
            for (int r = 0; r < 8; ++r) {
                int row = t * 16 + r + 8 * g;
                if (row < WW)
                    ow[((size_t)(bw * WW + row)) * CC + h * DD + nh * 16 + n] = (__bf16)o[r];
            }
        }
    }
}

// ---------------- kernel 6: output projection + inverse shift ----------------
// one wave computes a 32x64 tile, same blocking as gemm_qkv.

__global__ __launch_bounds__(32) void gemm_proj(const __bf16* __restrict__ ow,
                                                const __bf16* __restrict__ pT,
                                                const float* __restrict__ proj_b,
                                                const int* __restrict__ shiftp,
                                                float* __restrict__ out) {
    int ct4 = blockIdx.x;                 // 0..3   (64 cols each)
    int rt2 = blockIdx.y;                 // 0..1499 (32 rows each)
    int l = threadIdx.x, g = l >> 4, n = l & 15;
    const __bf16* ap0 = ow + (size_t)(rt2 * 32 + n) * CC;
    const __bf16* ap1 = ap0 + 16 * CC;
    const __bf16* bp  = pT + (size_t)(ct4 * 64 + n) * CC;

    v8f acc[2][4] = {};
#pragma unroll
    for (int kc = 0; kc < 8; ++kc) {
        v16bf a0 = load_afrag(ap0 + kc * 32, g);
        v16bf a1 = load_afrag(ap1 + kc * 32, g);
#pragma unroll
        for (int i = 0; i < 4; ++i) {
            v16bf b = *(const v16bf*)(bp + (size_t)i * 16 * CC + kc * 32 + 16 * g);
            acc[0][i] = wmma_bf16(a0, b, acc[0][i]);
            acc[1][i] = wmma_bf16(a1, b, acc[1][i]);
        }
    }

    int shift = shiftp[0];
#pragma unroll
    for (int i = 0; i < 4; ++i) {
        int c = ct4 * 64 + i * 16 + n;
        float bias = proj_b[c];
#pragma unroll
        for (int s = 0; s < 2; ++s)
#pragma unroll
            for (int r = 0; r < 8; ++r) {
                int row = rt2 * 32 + s * 16 + r + 8 * g;
                int b_ = row / LL, ts = row % LL;
                // inverse of the forward roll: shifted index ts -> final index tf
                int tf = (!shift || ts < WW) ? ts : WW + (ts - WW + WW / 2) % (LL - WW);
                out[((size_t)(b_ * LL + tf)) * CC + c] = acc[s][i][r] + bias;
            }
    }
}

// ---------------- launch ----------------

extern "C" void kernel_launch(void* const* d_in, const int* in_sizes, int n_in,
                              void* d_out, int out_size, void* d_ws, size_t ws_size,
                              hipStream_t stream) {
    const float* query  = (const float*)d_in[0];
    const float* qpos   = (const float*)d_in[1];
    const float* qk_w   = (const float*)d_in[2];
    const float* qk_b   = (const float*)d_in[3];
    const float* v_w    = (const float*)d_in[4];
    const float* v_b    = (const float*)d_in[5];
    const float* proj_w = (const float*)d_in[6];
    const float* proj_b = (const float*)d_in[7];
    const int*   shiftp = (const int*)d_in[8];

    char* ws = (char*)d_ws;
    const size_t SZ_X  = (size_t)ROWS * CC * 2;        // 24,576,000
    const size_t SZ_QK = (size_t)BW * HH * WP * DD * 2;// 24,903,680
    const size_t SZ_V  = (size_t)BW * HH * VP * DD * 2;// 26,214,400
    __bf16* xpos = (__bf16*)(ws);
    __bf16* xbf  = (__bf16*)(ws + SZ_X);
    __bf16* qw   = (__bf16*)(ws + 2 * SZ_X);
    __bf16* kw   = (__bf16*)(ws + 2 * SZ_X + SZ_QK);
    __bf16* vw   = (__bf16*)(ws + 2 * SZ_X + 2 * SZ_QK);
    __bf16* wT   = (__bf16*)(ws + 2 * SZ_X + 2 * SZ_QK + SZ_V);
    __bf16* pT   = (__bf16*)(ws + 2 * SZ_X + 2 * SZ_QK + SZ_V + 768 * CC * 2);
    __bf16* ow   = xpos;   // xpos is dead after gemm_qkv; reuse for attention output

    prep<<<ROWS * 64 / 256, 256, 0, stream>>>(query, qpos, shiftp, xpos, xbf);
    wprep<<<1024, 128, 0, stream>>>(qk_w, v_w, proj_w, wT, pT);
    padzero<<<BW * HH, 128, 0, stream>>>(qw, kw, vw);
    gemm_qkv<<<dim3(12, ROWS / 32), 32, 0, stream>>>(xpos, xbf, wT, qk_b, v_b, qw, kw, vw);
    attn_win<<<BW * HH, 128, 0, stream>>>(qw, kw, vw, ow);
    gemm_proj<<<dim3(4, ROWS / 32), 32, 0, stream>>>(ow, pT, proj_b, shiftp, (float*)d_out);
}
